// Str2MSA_18399639896109
// MI455X (gfx1250) — matrix operand order repaired and checked by reading, not compiled, not run on hardware
//
#include <hip/hip_runtime.h>

// ---------------------------------------------------------------------------
// Str2MSA on MI455X (gfx1250): bf16 WMMA for all GEMMs, fp32 accumulation.
// B=1, N=512, L=512, d_msa=64, d_state=32, H=4, DK=32, d_ff=256.
// Heavy GEMM stages its B tile into LDS with global_load_async_to_lds_b128.
// ---------------------------------------------------------------------------

typedef __attribute__((ext_vector_type(16))) __bf16 v16bf;
typedef __attribute__((ext_vector_type(8)))  float  v8f;

__device__ __forceinline__ unsigned short f2bf(float f) {
    union { float f; unsigned u; } v; v.f = f;
    unsigned r = v.u + 0x7FFFu + ((v.u >> 16) & 1u);   // round-to-nearest-even
    return (unsigned short)(r >> 16);
}

// A operand (16x32 bf16, M=lane row). ISA layout:
// lanes 0-15: K=0..7 (v0..3), K=16..23 (v4..7); lanes 16-31: K=8..15, 24..31.
__device__ __forceinline__ v16bf load_a(const unsigned short* rowbase, int kk, bool hi) {
    const unsigned* p = (const unsigned*)(rowbase + kk);   // bf16 pairs
    union { v16bf v; unsigned u[8]; } a;
    int o = hi ? 4 : 0;
#pragma unroll
    for (int i = 0; i < 4; ++i) { a.u[i] = p[o + i]; a.u[4 + i] = p[o + 8 + i]; }
    return a.v;
}

// B operand (32x16 bf16, N=lane col, column-major source: colbase = &B[col][0]).
// lanes 0-15: K=0..7, K=16..23; lanes 16-31: K=8..15, 24..31 (same as A).
__device__ __forceinline__ v16bf load_b(const unsigned short* colbase, int kk, bool hi) {
    const unsigned* p = (const unsigned*)(colbase + kk);
    union { v16bf v; unsigned u[8]; } b;
    int o = hi ? 4 : 0;
#pragma unroll
    for (int i = 0; i < 4; ++i) { b.u[i] = p[o + i]; b.u[4 + i] = p[o + 8 + i]; }
    return b.v;
}

__device__ __forceinline__ v8f wmma_bf16(v16bf a, v16bf b, v8f c) {
    return __builtin_amdgcn_wmma_f32_16x16x32_bf16(false, a, false, b, (short)0, c, false, false);
}

// ---------------------------------------------------------------------------
// Weight transpose fp32 -> bf16 column-major: dst[n][k] = src[k][n]
// ---------------------------------------------------------------------------
__global__ void k_transpose_bf16(const float* __restrict__ src,
                                 unsigned short* __restrict__ dst, int K, int N) {
    int idx = blockIdx.x * blockDim.x + threadIdx.x;
    if (idx >= K * N) return;
    int k = idx / N, n = idx % N;
    dst[n * K + k] = f2bf(src[k * N + n]);
}

// ---------------------------------------------------------------------------
// LN(state) then q = sn@wq+bq (scaled), k = sn@wk+bk. One block per position.
// ---------------------------------------------------------------------------
__global__ void __launch_bounds__(128)
k_state_qk(const float* __restrict__ state, const float* __restrict__ gs,
           const float* __restrict__ bs, const float* __restrict__ wq,
           const float* __restrict__ bq, const float* __restrict__ wk,
           const float* __restrict__ bk, float* __restrict__ qbuf,
           float* __restrict__ kbuf) {
    int l = blockIdx.x;
    int t = threadIdx.x;
    __shared__ float sn[32], snn[32];
    __shared__ float mu_s, rs_s;
    if (t < 32) sn[t] = state[l * 32 + t];
    __syncthreads();
    if (t == 0) {
        float mu = 0.f;
        for (int d = 0; d < 32; ++d) mu += sn[d];
        mu *= (1.f / 32.f);
        float var = 0.f;
        for (int d = 0; d < 32; ++d) { float x = sn[d] - mu; var += x * x; }
        var *= (1.f / 32.f);
        mu_s = mu; rs_s = rsqrtf(var + 1e-5f);
    }
    __syncthreads();
    if (t < 32) snn[t] = (sn[t] - mu_s) * rs_s * gs[t] + bs[t];
    __syncthreads();
    float aq = bq[t], ak = bk[t];
    for (int d = 0; d < 32; ++d) {
        aq += snn[d] * wq[d * 128 + t];
        ak += snn[d] * wk[d * 128 + t];
    }
    int h = t >> 5, dk = t & 31;
    qbuf[(h * 512 + l) * 32 + dk] = aq * 0.17677669529663687f;  // 1/sqrt(32)
    kbuf[(h * 512 + l) * 32 + dk] = ak;
}

// ---------------------------------------------------------------------------
// attn = softmax(mask(q . k^T)); one block per (h,i) row; write bf16 probs.
// ---------------------------------------------------------------------------
__global__ void __launch_bounds__(256)
k_attn_softmax(const float* __restrict__ qbuf, const float* __restrict__ kbuf,
               const float* __restrict__ xyz, unsigned short* __restrict__ attn) {
    int hi_ = blockIdx.x;                   // h*512 + i
    int h = hi_ >> 9, i = hi_ & 511;
    int t = threadIdx.x;
    __shared__ float qr[32];
    __shared__ float cai[3];
    __shared__ float red[256];
    if (t < 32) qr[t] = qbuf[(h * 512 + i) * 32 + t];
    if (t < 3)  cai[t] = xyz[(i * 3 + 1) * 3 + t];
    __syncthreads();
    float db = (h == 0) ? 8.f : (h == 1) ? 12.f : (h == 2) ? 16.f : 20.f;
    float db2 = db * db;
    float lg[2];
#pragma unroll
    for (int s = 0; s < 2; ++s) {
        int j = t + s * 256;
        const float* kr = kbuf + (h * 512 + j) * 32;
        float dot = 0.f;
        for (int d = 0; d < 32; ++d) dot += qr[d] * kr[d];
        float dx = cai[0] - xyz[(j * 3 + 1) * 3 + 0];
        float dy = cai[1] - xyz[(j * 3 + 1) * 3 + 1];
        float dz = cai[2] - xyz[(j * 3 + 1) * 3 + 2];
        float sq = dx * dx + dy * dy + dz * dz;      // dist > db  <=>  sq > db^2
        lg[s] = (sq > db2) ? -1e30f : dot;
    }
    float m = fmaxf(lg[0], lg[1]);
    red[t] = m; __syncthreads();
    for (int s2 = 128; s2 > 0; s2 >>= 1) {
        if (t < s2) red[t] = fmaxf(red[t], red[t + s2]);
        __syncthreads();
    }
    m = red[0]; __syncthreads();
    float e0 = __expf(lg[0] - m), e1 = __expf(lg[1] - m);
    red[t] = e0 + e1; __syncthreads();
    for (int s2 = 128; s2 > 0; s2 >>= 1) {
        if (t < s2) red[t] += red[t + s2];
        __syncthreads();
    }
    float inv = 1.f / red[0];
    size_t base = (size_t)(h * 512 + i) * 512;
    attn[base + t]       = f2bf(e0 * inv);
    attn[base + t + 256] = f2bf(e1 * inv);
}

// ---------------------------------------------------------------------------
// Row LayerNorm over 64 channels, fp32 in -> bf16 out. One block per row.
// ---------------------------------------------------------------------------
__global__ void __launch_bounds__(64)
k_ln_row64(const float* __restrict__ in, const float* __restrict__ g,
           const float* __restrict__ b, unsigned short* __restrict__ outb) {
    size_t row = blockIdx.x;
    int t = threadIdx.x;
    __shared__ float red[64];
    float v = in[row * 64 + t];
    red[t] = v; __syncthreads();
    for (int s = 32; s > 0; s >>= 1) { if (t < s) red[t] += red[t + s]; __syncthreads(); }
    float mu = red[0] * (1.f / 64.f); __syncthreads();
    float d = v - mu;
    red[t] = d * d; __syncthreads();
    for (int s = 32; s > 0; s >>= 1) { if (t < s) red[t] += red[t + s]; __syncthreads(); }
    float rs = rsqrtf(red[0] * (1.f / 64.f) + 1e-5f);
    outb[row * 64 + t] = f2bf(d * rs * g[t] + b[t]);
}

// ---------------------------------------------------------------------------
// V = LN(msa) @ wv + bv, written transposed: Vt[n][h][c][j]  (K=j contiguous).
// One wave per 16-row tile; 2 K-steps x 4 N-tiles = 8 WMMA.
// D rows map to consecutive j -> pack each lane's 8 results into one b128.
// ---------------------------------------------------------------------------
__global__ void __launch_bounds__(32)
k_gemm_v(const unsigned short* __restrict__ mln, const unsigned short* __restrict__ wvT,
         const float* __restrict__ bv, unsigned short* __restrict__ Vt) {
    int r0 = blockIdx.x * 16;
    int n  = r0 >> 9;
    int j0 = r0 & 511;
    int lane = threadIdx.x, lrow = lane & 15;
    bool hi = lane >= 16;
    const unsigned short* Arow = mln + (size_t)(r0 + lrow) * 64;
    v8f acc[4] = {};
#pragma unroll
    for (int kk = 0; kk < 64; kk += 32) {
        v16bf a = load_a(Arow, kk, hi);
#pragma unroll
        for (int nt = 0; nt < 4; ++nt)
            acc[nt] = wmma_bf16(a, load_b(wvT + (nt * 16 + lrow) * 64, kk, hi), acc[nt]);
    }
#pragma unroll
    for (int nt = 0; nt < 4; ++nt) {
        int c = nt * 16 + lrow;
        float bias = bv[c];
        size_t base = (((size_t)n * 4 + nt) * 16 + lrow) * 512 + j0 + (hi ? 8 : 0);
        unsigned pk[4];
#pragma unroll
        for (int p = 0; p < 4; ++p)
            pk[p] = (unsigned)f2bf(acc[nt][2 * p] + bias) |
                    ((unsigned)f2bf(acc[nt][2 * p + 1] + bias) << 16);
        uint4 w; w.x = pk[0]; w.y = pk[1]; w.z = pk[2]; w.w = pk[3];
        *(uint4*)(Vt + base) = w;                 // 16B aligned: j0%16==0
    }
}

// ---------------------------------------------------------------------------
// The heavy GEMM: out[n,i,16h:16h+16] = sum_j attn[h,i,j] * Vt[n,h,:,j].
// One 256-thread block (8 waves) per (n,h). The 16x512 bf16 B tile (16 KB) is
// staged into LDS ONCE via global_load_async_to_lds_b128 (ASYNCcnt path) and
// reused by all 32 i-tiles (was re-read 32x from L2 before). Each wave then
// runs 4 i-tiles x 16 K-step WMMAs with A from global, B from LDS.
// ---------------------------------------------------------------------------
__global__ void __launch_bounds__(256)
k_attn_apply(const unsigned short* __restrict__ attn, const unsigned short* __restrict__ Vt,
             unsigned short* __restrict__ aout) {
    __shared__ unsigned short vtile[16 * 512];           // 16 KB B tile for (n,h)
    int id = blockIdx.x;                                  // n*4 + h
    int h = id & 3, n = id >> 2;
    int t = threadIdx.x;

    // Async-stage Vt[n][h] (16 rows x 512 bf16) into LDS: 1024 x 16-byte chunks,
    // 256 lanes x 4 chunks. Each lane supplies its own LDS + global address.
    const unsigned short* src = Vt + (((size_t)n * 4 + h) * 16) * 512;
#pragma unroll
    for (int c = 0; c < 4; ++c) {
        int elt = (t + c * 256) * 8;                      // 8 bf16 = 16 bytes
        unsigned lds = (unsigned)(uintptr_t)(&vtile[elt]); // low 32b = LDS addr
        const unsigned short* g = src + elt;
        asm volatile("global_load_async_to_lds_b128 %0, %1, off"
                     :: "v"(lds), "v"(g) : "memory");
    }
    asm volatile("s_wait_asynccnt 0x0" ::: "memory");
    __syncthreads();

    int wave = t >> 5;                                    // wave-uniform
    int lane = t & 31, lrow = lane & 15;
    bool hi = lane >= 16;
    const unsigned short* Bcol = vtile + lrow * 512;      // ds_load B operand
#pragma unroll 1
    for (int sub = 0; sub < 4; ++sub) {
        int i0 = (wave + sub * 8) * 16;
        const unsigned short* Arow = attn + (size_t)(h * 512 + i0 + lrow) * 512;
        v8f acc = {};
        for (int jj = 0; jj < 512; jj += 32) {
            __builtin_prefetch(Arow + jj + 128, 0, 1);    // global_prefetch_b8
            v16bf a = load_a(Arow, jj, hi);
            v16bf b = load_b(Bcol, jj, hi);
            acc = wmma_bf16(a, b, acc);
        }
        int ccol = h * 16 + lrow;
#pragma unroll
        for (int r = 0; r < 8; ++r) {
            int i = i0 + r + (hi ? 8 : 0);
            aout[((size_t)n * 512 + i) * 64 + ccol] = f2bf(acc[r]);
        }
    }
}

// ---------------------------------------------------------------------------
// msa_res = msa + aout @ wo + bo  (fp32 into d_out, which doubles as residual)
// ---------------------------------------------------------------------------
__global__ void __launch_bounds__(32)
k_gemm_wo(const unsigned short* __restrict__ aout, const unsigned short* __restrict__ woT,
          const float* __restrict__ bo, const float* __restrict__ msa,
          float* __restrict__ out) {
    int r0 = blockIdx.x * 16;
    int lane = threadIdx.x, lrow = lane & 15;
    bool hi = lane >= 16;
    const unsigned short* Arow = aout + (size_t)(r0 + lrow) * 64;
    v8f acc[4] = {};
#pragma unroll
    for (int kk = 0; kk < 64; kk += 32) {
        v16bf a = load_a(Arow, kk, hi);
#pragma unroll
        for (int nt = 0; nt < 4; ++nt)
            acc[nt] = wmma_bf16(a, load_b(woT + (nt * 16 + lrow) * 64, kk, hi), acc[nt]);
    }
#pragma unroll
    for (int nt = 0; nt < 4; ++nt) {
        int c = nt * 16 + lrow;
        float bias = bo[c];
#pragma unroll
        for (int r = 0; r < 8; ++r) {
            size_t idx = (size_t)(r0 + r + (hi ? 8 : 0)) * 64 + c;
            out[idx] = msa[idx] + acc[nt][r] + bias;
        }
    }
}

// ---------------------------------------------------------------------------
// Fused FFN: out += relu(LN(msa_res) @ w1 + b1) @ w2 + b2.
// fc1: 16 N-tiles x 2 K-steps (32 wmma). D-layout -> A-layout via 8 KB LDS.
// fc2: 4 N-tiles x 8 K-steps (32 wmma). Residual add into d_out.
// ---------------------------------------------------------------------------
__global__ void __launch_bounds__(32)
k_ffn(const unsigned short* __restrict__ mln2, const unsigned short* __restrict__ w1T,
      const float* __restrict__ b1, const unsigned short* __restrict__ w2T,
      const float* __restrict__ b2, float* __restrict__ out) {
    __shared__ unsigned short hid[16 * 256];
    int r0 = blockIdx.x * 16;
    int lane = threadIdx.x, lrow = lane & 15;
    bool hi = lane >= 16;
    const unsigned short* Arow = mln2 + (size_t)(r0 + lrow) * 64;

    v8f acc1[16] = {};
#pragma unroll
    for (int kk = 0; kk < 64; kk += 32) {
        v16bf a = load_a(Arow, kk, hi);
#pragma unroll
        for (int nt = 0; nt < 16; ++nt)
            acc1[nt] = wmma_bf16(a, load_b(w1T + (nt * 16 + lrow) * 64, kk, hi), acc1[nt]);
    }
#pragma unroll
    for (int nt = 0; nt < 16; ++nt) {
        int c = nt * 16 + lrow;
        float bias = b1[c];
#pragma unroll
        for (int r = 0; r < 8; ++r) {
            float v = acc1[nt][r] + bias;
            v = v > 0.f ? v : 0.f;                        // ReLU
            hid[(r + (hi ? 8 : 0)) * 256 + c] = f2bf(v);  // D-layout -> row-major LDS
        }
    }
    __syncthreads();

    v8f acc2[4] = {};
#pragma unroll
    for (int kk = 0; kk < 256; kk += 32) {
        v16bf a = load_a(hid + lrow * 256, kk, hi);       // A-layout reads from LDS
#pragma unroll
        for (int nt = 0; nt < 4; ++nt)
            acc2[nt] = wmma_bf16(a, load_b(w2T + (nt * 16 + lrow) * 256, kk, hi), acc2[nt]);
    }
#pragma unroll
    for (int nt = 0; nt < 4; ++nt) {
        int c = nt * 16 + lrow;
        float bias = b2[c];
#pragma unroll
        for (int r = 0; r < 8; ++r) {
            size_t idx = (size_t)(r0 + r + (hi ? 8 : 0)) * 64 + c;
            out[idx] = out[idx] + acc2[nt][r] + bias;     // residual add
        }
    }
}

// ---------------------------------------------------------------------------
// Launch
// ---------------------------------------------------------------------------
extern "C" void kernel_launch(void* const* d_in, const int* in_sizes, int n_in,
                              void* d_out, int out_size, void* d_ws, size_t ws_size,
                              hipStream_t stream) {
    (void)in_sizes; (void)n_in; (void)out_size; (void)ws_size;
    const float* msa     = (const float*)d_in[0];
    const float* xyz     = (const float*)d_in[1];
    const float* state   = (const float*)d_in[2];
    const float* gamma_s = (const float*)d_in[3];
    const float* beta_s  = (const float*)d_in[4];
    const float* gamma1  = (const float*)d_in[5];
    const float* beta1   = (const float*)d_in[6];
    const float* gamma2  = (const float*)d_in[7];
    const float* beta2   = (const float*)d_in[8];
    const float* wq = (const float*)d_in[9];
    const float* bq = (const float*)d_in[10];
    const float* wk = (const float*)d_in[11];
    const float* bk = (const float*)d_in[12];
    const float* wv = (const float*)d_in[13];
    const float* bv = (const float*)d_in[14];
    const float* wo = (const float*)d_in[15];
    const float* bo = (const float*)d_in[16];
    const float* w1 = (const float*)d_in[17];
    const float* b1 = (const float*)d_in[18];
    const float* w2 = (const float*)d_in[19];
    const float* b2 = (const float*)d_in[20];

    char* ws = (char*)d_ws;
    float* qbuf          = (float*)(ws + 0);                       // 256 KB
    float* kbuf          = (float*)(ws + 262144);                  // 256 KB
    unsigned short* attn = (unsigned short*)(ws + 524288);         // 2 MB bf16
    unsigned short* wvT  = (unsigned short*)(ws + 2621440);        // 8 KB
    unsigned short* woT  = (unsigned short*)(ws + 2629632);        // 8 KB
    unsigned short* w1T  = (unsigned short*)(ws + 2637824);        // 32 KB
    unsigned short* w2T  = (unsigned short*)(ws + 2670592);        // 32 KB
    unsigned short* mln  = (unsigned short*)(ws + 2703360);        // 32 MB (LN1, reused LN2)
    unsigned short* Vt   = (unsigned short*)(ws + 2703360 + 33554432);      // 32 MB
    unsigned short* aout = (unsigned short*)(ws + 2703360 + 2 * 33554432);  // 32 MB
    float* out = (float*)d_out;                                    // also fp32 residual

    k_transpose_bf16<<<(64 * 64 + 255) / 256, 256, 0, stream>>>(wv, wvT, 64, 64);
    k_transpose_bf16<<<(64 * 64 + 255) / 256, 256, 0, stream>>>(wo, woT, 64, 64);
    k_transpose_bf16<<<(64 * 256 + 255) / 256, 256, 0, stream>>>(w1, w1T, 64, 256);
    k_transpose_bf16<<<(256 * 64 + 255) / 256, 256, 0, stream>>>(w2, w2T, 256, 64);

    k_state_qk<<<512, 128, 0, stream>>>(state, gamma_s, beta_s, wq, bq, wk, bk, qbuf, kbuf);
    k_attn_softmax<<<2048, 256, 0, stream>>>(qbuf, kbuf, xyz, attn);
    k_ln_row64<<<262144, 64, 0, stream>>>(msa, gamma1, beta1, mln);
    k_gemm_v<<<16384, 32, 0, stream>>>(mln, wvT, bv, Vt);
    k_attn_apply<<<2048, 256, 0, stream>>>(attn, Vt, aout);
    k_gemm_wo<<<16384, 32, 0, stream>>>(aout, woT, bo, msa, out);
    k_ln_row64<<<262144, 64, 0, stream>>>(out, gamma2, beta2, mln);
    k_ffn<<<16384, 32, 0, stream>>>(mln, w1T, b1, w2T, b2, out);
}
